// LSTMForARLM_83073257439739
// MI455X (gfx1250) — compile-verified, matrix operand order checked by reading, not run on hardware
//
#include <hip/hip_runtime.h>
#include <hip/hip_bf16.h>

// ---------------------------------------------------------------------------
// LSTM AR-LM forward for MI455X (gfx1250, wave32, WMMA).
//   emb-gather -> [2x](xg GEMM -> sequential LSTM scan) -> vocab GEMM
// bf16 operands, f32 accumulate via v_wmma_f32_16x16x32_bf16.
// GEMM B-operands staged block-wide in LDS with async-to-LDS copies
// (double-buffered). Scan k-loop kept un-unrolled + sched_barrier-fenced so
// the scheduler cannot hoist all Whh loads and spill (seen in prior rounds).
// ---------------------------------------------------------------------------

typedef __attribute__((ext_vector_type(16))) __bf16 v16bf;
typedef __attribute__((ext_vector_type(8)))  float  v8f;
typedef int v4i __attribute__((vector_size(16)));

#define WMMA_BF16(a, b, c) \
  __builtin_amdgcn_wmma_f32_16x16x32_bf16(false, (a), false, (b), (short)0, (c), false, false)

static constexpr int B    = 16;
static constexpr int T    = 256;
static constexpr int E    = 512;    // == H
static constexpr int H    = 512;
static constexpr int G4   = 4 * H;  // 2048
static constexpr int V    = 32000;
static constexpr int NTOK = B * T;  // 4096
static constexpr int SLABW = 40;    // padded LDS row stride (shorts): conflict-free

#if __has_builtin(__builtin_amdgcn_global_load_async_to_lds_b128)
#define STAGE_ASYNC 1
#endif

__device__ __forceinline__ unsigned short f2bf(float f) {
  unsigned int u = __float_as_uint(f);
  unsigned int r = u + 0x7FFFu + ((u >> 16) & 1u);   // round-to-nearest-even
  return (unsigned short)(r >> 16);
}
__device__ __forceinline__ float sigmoidf_(float x) { return 1.0f / (1.0f + __expf(-x)); }

// hard scheduling fence (keeps register pressure bounded)
__device__ __forceinline__ void sched_fence() {
#if __has_builtin(__builtin_amdgcn_sched_barrier)
  __builtin_amdgcn_sched_barrier(0);
#else
  asm volatile("" ::: "memory");
#endif
}

// stage 16 bytes global -> LDS (async if available)
__device__ __forceinline__ void stage16(const unsigned short* src, unsigned short* dst) {
#ifdef STAGE_ASYNC
  __builtin_amdgcn_global_load_async_to_lds_b128(
      (__attribute__((address_space(1))) v4i*)const_cast<unsigned short*>(src),
      (__attribute__((address_space(3))) v4i*)dst, 0, 0);
#else
  *(uint4*)dst = *(const uint4*)src;
#endif
}
__device__ __forceinline__ void wait_stage() {
#ifdef STAGE_ASYNC
#if __has_builtin(__builtin_amdgcn_s_wait_asynccnt)
  __builtin_amdgcn_s_wait_asynccnt(0);
#else
  asm volatile("s_wait_asynccnt 0x0" ::: "memory");
#endif
#endif
}

// --------------------------- f32 -> bf16 convert ---------------------------
__global__ void k_f32_to_bf16(const float* __restrict__ src,
                              unsigned short* __restrict__ dst, int n) {
  int stride = gridDim.x * blockDim.x;
  for (int i = blockIdx.x * blockDim.x + threadIdx.x; i < n; i += stride)
    dst[i] = f2bf(src[i]);
}

// ----------------------------- bias pre-sum --------------------------------
__global__ void k_bias_sum(const float* __restrict__ a, const float* __restrict__ b,
                           float* __restrict__ s, int n) {
  int i = blockIdx.x * blockDim.x + threadIdx.x;
  if (i < n) s[i] = a[i] + b[i];
}

// ------------------------ embedding gather (+pad mask) ---------------------
__global__ void k_embed(const int* __restrict__ ids,
                        const float* __restrict__ emb,
                        unsigned short* __restrict__ xb) {
  int stride = gridDim.x * blockDim.x;
  for (int i = blockIdx.x * blockDim.x + threadIdx.x; i < NTOK * E; i += stride) {
    int n = i >> 9, d = i & (E - 1);
    int id = ids[n];
    float v = (id == 0) ? 0.0f : emb[id * E + d];
    xb[i] = f2bf(v);
  }
}

// ------------------- generic 16xN WMMA GEMM: OUT = X@W^T + bias ------------
// block = 8 waves sharing one 64-column N-block (B slab staged in LDS,
// double buffered); wave w owns M-tile (mblock*8+w). grid = 32 * (ncols/64).
// REMAP: store to XG layout [t][col][b] (f32) consumed by the LSTM scan.
template <bool REMAP>
__global__ void __launch_bounds__(256, 1)
k_gemm16(const unsigned short* __restrict__ X,   // [M, 512] bf16
         const unsigned short* __restrict__ W,   // [ncols, 512] bf16
         const float* __restrict__ bias,         // [ncols]
         float* __restrict__ OUT,                // REMAP? [T][G4][16] : [M, ncols]
         long ncols)
{
  __shared__ __align__(16) unsigned short slab[2][64 * SLABW];  // 10 KB
  const int tid   = threadIdx.x;
  const int lane  = tid & 31;
  const int wave  = tid >> 5;
  const int l15   = lane & 15;
  const int lhalf = lane >> 4;
  const int mtile   = (blockIdx.x & 31) * 8 + wave;
  const int nbase16 = (blockIdx.x >> 5) * 64;       // first of 64 columns

  const unsigned short* Arow = X + (mtile * 16 + l15) * 512 + lhalf * 16;

  // cooperative slab staging: 256 threads x 16B = 64 rows x 32 k (4 KB)
  const int srow = tid >> 2, sseg = tid & 3;
  const unsigned short* Wrow = W + (size_t)(nbase16 + srow) * 512 + sseg * 8;
  unsigned short* Sdst = &slab[0][0] + srow * SLABW + sseg * 8;

  // prologue: stage k=0, load A frag k=0
  stage16(Wrow, Sdst);
  v16bf a = *(const v16bf*)(Arow);
  v16bf aNext = a;
  wait_stage();
  __syncthreads();

  v8f acc[4];
#pragma unroll
  for (int i = 0; i < 4; ++i) acc[i] = (v8f){0, 0, 0, 0, 0, 0, 0, 0};

#pragma unroll
  for (int k = 0; k < 16; ++k) {                  // K = 512 in steps of 32
    if (k < 15) {                                 // issue next-slab + next-A early
      stage16(Wrow + (k + 1) * 32, Sdst + ((k + 1) & 1) * 64 * SLABW);
      aNext = *(const v16bf*)(Arow + (k + 1) * 32);
    }
    const unsigned short* sb = &slab[k & 1][0];
#pragma unroll
    for (int nt = 0; nt < 4; ++nt) {
      v16bf b = *(const v16bf*)(sb + (nt * 16 + l15) * SLABW + lhalf * 16);
      acc[nt] = WMMA_BF16(a, b, acc[nt]);
    }
    a = aNext;
    wait_stage();        // next slab landed
    __syncthreads();     // all waves done reading current slab
  }

#pragma unroll
  for (int nt = 0; nt < 4; ++nt) {
    int col = nbase16 + nt * 16 + l15;
    float bs = bias[col];
#pragma unroll
    for (int e = 0; e < 8; ++e) {
      int m = mtile * 16 + e + 8 * lhalf;          // token n = b*T + t
      long idx;
      if (REMAP) {
        int tt = m & (T - 1), bb = m >> 8;
        idx = ((long)tt * G4 + col) * 16 + bb;     // [t][col][b]
      } else {
        idx = (long)m * ncols + col;
      }
      OUT[idx] = acc[nt][e] + bs;
    }
  }
}

// --------------------------- sequential LSTM scan --------------------------
// One persistent workgroup, 16 waves. Wave w owns hidden slice [32w, 32w+32)
// across all 4 gates (8 acc tiles) -> gate fusion is wave-local.
// h (bf16) + c (f32) in LDS. k-loop NOT unrolled + sched fences: max 4 Whh
// fragments in flight -> ~115 VGPRs, no scratch spills.
__global__ void __launch_bounds__(512, 1)
k_lstm_scan(const float* __restrict__ XG,            // [T][G4][16] f32
            const unsigned short* __restrict__ Whh,  // [G4, H] bf16
            unsigned short* __restrict__ Xout)       // [NTOK, H] bf16 (n-order)
{
  __shared__ __align__(32) unsigned short h_lds[B * H]; // 16 KB
  __shared__ float c_lds[B * H];                        // 32 KB
  const int tid   = threadIdx.x;
  const int lane  = tid & 31;
  const int wave  = tid >> 5;
  const int l15   = lane & 15;
  const int lhalf = lane >> 4;

  for (int i = tid; i < B * H; i += 512) { h_lds[i] = 0; c_lds[i] = 0.0f; }
  __syncthreads();

  const unsigned short* Wbase = Whh + (size_t)(wave * 32 + l15) * H + lhalf * 16;

#pragma unroll 1
  for (int t = 0; t < T; ++t) {
    // seed accumulators with xg[t]: one v8f (32B) load per tile
    v8f acc[8];                                  // acc[g*2+nt]
#pragma unroll
    for (int g = 0; g < 4; ++g)
#pragma unroll
      for (int nt = 0; nt < 2; ++nt) {
        int col = g * H + wave * 32 + nt * 16 + l15;
        acc[g * 2 + nt] = *(const v8f*)(XG + ((size_t)t * G4 + col) * 16 + lhalf * 8);
      }

#pragma unroll 1                                 // keep pressure bounded per k
    for (int k = 0; k < 16; ++k) {
      v16bf a = *(const v16bf*)(h_lds + l15 * H + k * 32 + lhalf * 16);
      {
        v16bf b0 = *(const v16bf*)(Wbase + (size_t)(0 * H +  0) * H + k * 32);
        v16bf b1 = *(const v16bf*)(Wbase + (size_t)(0 * H + 16) * H + k * 32);
        v16bf b2 = *(const v16bf*)(Wbase + (size_t)(1 * H +  0) * H + k * 32);
        v16bf b3 = *(const v16bf*)(Wbase + (size_t)(1 * H + 16) * H + k * 32);
        acc[0] = WMMA_BF16(a, b0, acc[0]);
        acc[1] = WMMA_BF16(a, b1, acc[1]);
        acc[2] = WMMA_BF16(a, b2, acc[2]);
        acc[3] = WMMA_BF16(a, b3, acc[3]);
      }
      sched_fence();                             // second half stays second
      {
        v16bf b0 = *(const v16bf*)(Wbase + (size_t)(2 * H +  0) * H + k * 32);
        v16bf b1 = *(const v16bf*)(Wbase + (size_t)(2 * H + 16) * H + k * 32);
        v16bf b2 = *(const v16bf*)(Wbase + (size_t)(3 * H +  0) * H + k * 32);
        v16bf b3 = *(const v16bf*)(Wbase + (size_t)(3 * H + 16) * H + k * 32);
        acc[4] = WMMA_BF16(a, b0, acc[4]);
        acc[5] = WMMA_BF16(a, b1, acc[5]);
        acc[6] = WMMA_BF16(a, b2, acc[6]);
        acc[7] = WMMA_BF16(a, b3, acc[7]);
      }
    }
    __syncthreads();   // all reads of h done before anyone rewrites it

    // wave-local gate fusion: i,f,g,o -> c,h
#pragma unroll
    for (int nt = 0; nt < 2; ++nt) {
      int j = wave * 32 + nt * 16 + l15;
#pragma unroll
      for (int e = 0; e < 8; ++e) {
        int bb = e + 8 * lhalf;
        float iv = sigmoidf_(acc[nt][e]);
        float fv = sigmoidf_(acc[2 + nt][e]);
        float gv = tanhf(acc[4 + nt][e]);
        float ov = sigmoidf_(acc[6 + nt][e]);
        float c  = fv * c_lds[bb * H + j] + iv * gv;
        c_lds[bb * H + j] = c;
        unsigned short hb = f2bf(ov * tanhf(c));
        h_lds[bb * H + j] = hb;
        Xout[(bb * T + t) * H + j] = hb;
      }
    }
    __syncthreads();   // h update visible before next step's reads
  }
}

// ---------------------------------------------------------------------------
extern "C" void kernel_launch(void* const* d_in, const int* in_sizes, int n_in,
                              void* d_out, int out_size, void* d_ws, size_t ws_size,
                              hipStream_t stream) {
  const int*   ids  = (const int*)  d_in[0];
  const float* emb  = (const float*)d_in[1];
  const float* Wih  = (const float*)d_in[2];   // [2, 2048, 512]
  const float* Whh  = (const float*)d_in[3];   // [2, 2048, 512]
  const float* bih  = (const float*)d_in[4];   // [2, 2048]
  const float* bhh  = (const float*)d_in[5];   // [2, 2048]
  const float* Wout = (const float*)d_in[6];   // [32000, 512]
  const float* bout = (const float*)d_in[7];   // [32000]
  float* out = (float*)d_out;

  char* ws = (char*)d_ws;
  size_t off = 0;
  unsigned short* Wih_b  = (unsigned short*)(ws + off); off += (size_t)2 * G4 * E * 2;  // 4 MB
  unsigned short* Whh_b  = (unsigned short*)(ws + off); off += (size_t)2 * G4 * H * 2;  // 4 MB
  unsigned short* Wout_b = (unsigned short*)(ws + off); off += (size_t)V * H * 2;       // 32.8 MB
  unsigned short* xbuf   = (unsigned short*)(ws + off); off += (size_t)NTOK * E * 2;    // 4 MB
  float*          xg     = (float*)(ws + off);          off += (size_t)NTOK * G4 * 4;   // 33.6 MB
  unsigned short* hbuf   = (unsigned short*)(ws + off); off += (size_t)NTOK * H * 2;    // 4 MB
  float*          bsum   = (float*)(ws + off);          off += (size_t)2 * G4 * 4;      // 16 KB
  (void)off; (void)ws_size; (void)in_sizes; (void)n_in; (void)out_size;

  // weight conversions (f32 -> bf16) + bias pre-sum
  k_f32_to_bf16<<<2048, 256, 0, stream>>>(Wih,  Wih_b,  2 * G4 * E);
  k_f32_to_bf16<<<2048, 256, 0, stream>>>(Whh,  Whh_b,  2 * G4 * H);
  k_f32_to_bf16<<<4096, 256, 0, stream>>>(Wout, Wout_b, V * H);
  k_bias_sum<<<16, 256, 0, stream>>>(bih, bhh, bsum, 2 * G4);

  // embedding gather + pad mask
  k_embed<<<2048, 256, 0, stream>>>(ids, emb, xbuf);

  // layer 0
  k_gemm16<true><<<32 * (G4 / 64), 256, 0, stream>>>(xbuf, Wih_b, bsum, xg, (long)G4);
  k_lstm_scan<<<1, 512, 0, stream>>>(xg, Whh_b, hbuf);
  // layer 1
  k_gemm16<true><<<32 * (G4 / 64), 256, 0, stream>>>(hbuf, Wih_b + (size_t)G4 * E,
                                                     bsum + G4, xg, (long)G4);
  k_lstm_scan<<<1, 512, 0, stream>>>(xg, Whh_b + (size_t)G4 * H, xbuf);

  // vocab projection (dominant kernel: 134 GFLOP, 524 MB store-bound)
  k_gemm16<false><<<32 * (V / 64), 256, 0, stream>>>(xbuf, Wout_b, bout, out, (long)V);
}
